// LatentRNN_63574105915482
// MI455X (gfx1250) — matrix operand order repaired and checked by reading, not compiled, not run on hardware
//
#include <hip/hip_runtime.h>
#include <hip/hip_bf16.h>

typedef __attribute__((ext_vector_type(16))) _Float16 v16h;
typedef __attribute__((ext_vector_type(8)))  _Float16 v8h;
typedef __attribute__((ext_vector_type(8)))  float    v8f;

// ---------------- problem dims ----------------
#define BB 2048
#define TT 512
#define PP 8
#define LL 64
#define HH 180

// ---------------- tiling ----------------
#define NT_H 12    // 192/16  (H padded to 192)
#define NT_L 4     // 64/16
#define KT_X 4     // K=128   ([phys8 | cur64 | pad] -> 128)
#define KT_HP 2    // K=64    (latents)
#define KT_H 6     // K=192   (H padded)
#define KT_O1 8    // K=256   ([h192 | cur64])
#define FR 512     // halves per B-fragment (32 lanes * 16 halves = 32x16 tile)

// packed-weight fragment offsets (in halves) inside d_ws
#define OFF_WIN 0
#define OFF_WHP (OFF_WIN + NT_H*KT_X*FR)
#define OFF_WIH (OFF_WHP + NT_H*KT_HP*FR)
#define OFF_WHH (OFF_WIH + 3*NT_H*KT_H*FR)
#define OFF_WO1 (OFF_WHH + 3*NT_H*KT_H*FR)
#define OFF_WO2 (OFF_WO1 + NT_H*KT_O1*FR)
#define FRAG_TOTAL (OFF_WO2 + NT_L*KT_H*FR)

// padded bias region (floats), placed after the fragments
#define BO_IN 0
#define BO_HP 192
#define BO_IH 384
#define BO_HH 960
#define BO_O1 1536
#define BO_O2 1728
#define BIAS_TOTAL 1792

// per-wave LDS strides (in elements); chosen 16B-aligned & bank-spread
#define SR_XIN 136
#define SR_ACT 200
#define SR_CURH 72
#define SR_CURF 68

// ---------------- helpers ----------------
__device__ __forceinline__ float gelu_erf(float x) {
    return 0.5f * x * (1.0f + erff(x * 0.70710678118654752f));
}
__device__ __forceinline__ float sigm(float x) {
    return 1.0f / (1.0f + __expf(-x));
}
__device__ __forceinline__ v8f bcast8(float v) {
    v8f r;
#pragma unroll
    for (int i = 0; i < 8; ++i) r[i] = v;
    return r;
}

// WMMA D = A(16x32 f16) * B(32x16 f16) + C(16x16 f32)
__device__ __forceinline__ v8f wmma_f16(v16h a, v16h b, v8f c) {
    return __builtin_amdgcn_wmma_f32_16x16x32_f16(false, a, false, b, (short)0, c,
                                                  false, false);
}

// Load an A fragment (16x32, f16) from a row-major LDS half buffer.
// ISA A-layout: lanes 0-15 row m=lane, VGPR0..3 hold K=0..7, VGPR4..7 hold K=16..23;
// lanes 16-31 hold K=8..15 and K=24..31.  => two contiguous 16B chunks per lane.
__device__ __forceinline__ v16h load_a(const _Float16* buf, int strideH, int kbase,
                                       int lane) {
    int row = lane & 15;
    int grp = lane >> 4;
    const _Float16* p = buf + row * strideH + kbase + 8 * grp;
    v8h lo = *(const v8h*)(p);       // K = kbase + 8g .. +7
    v8h hi = *(const v8h*)(p + 16);  // K = kbase + 16 + 8g ..
    v16h a;
#pragma unroll
    for (int i = 0; i < 8; ++i) { a[i] = lo[i]; a[i + 8] = hi[i]; }
    return a;
}

// B fragments are pre-packed contiguously: lane*16 halves each (32B).
__device__ __forceinline__ v16h load_b(const _Float16* fragbase, int lane) {
    return *(const v16h*)(fragbase + lane * 16);
}

// ---------------- weight pre-pack kernel ----------------
// Packs W [N x Ksrc] (row-major f32) into WMMA B (K x N = W^T) fragment layout, f16.
// B element (k,n) of tile (nt,kt): lane = (n%16) + 16*((k%32 / 8) & 1), half j = (k & 7) + 8*(k%32 >= 16).
// n grouping (nGroup>0) maps padded gate blocks; k split/resume maps W_o1's
// [h | pad | cur] column gap.
__global__ void lrnn_pack_w(const float* __restrict__ src, _Float16* __restrict__ dst,
                            int NT, int KT, int Ksrc, int nGroup, int nReal,
                            int kSplit, int kResume, int kReal) {
    int total = NT * KT * FR;
    for (int idx = blockIdx.x * blockDim.x + threadIdx.x; idx < total;
         idx += gridDim.x * blockDim.x) {
        int t = idx / FR, r = idx % FR;
        int lane = r / 16, j = r % 16;
        int nt = t / KT, kt = t % KT;
        int nn = lane & 15, kkhi = lane >> 4;
        int kk = (j < 8) ? (kkhi * 8 + j) : (16 + kkhi * 8 + (j - 8));
        int n = nt * 16 + nn;
        int k = kt * 32 + kk;
        int srow; bool nok;
        if (nGroup) { int g = n / nGroup, m = n % nGroup; nok = (m < nReal); srow = g * nReal + m; }
        else        { nok = (n < nReal); srow = n; }
        int sk; bool kok;
        if (k < kSplit)        { sk = k; kok = true; }
        else if (k >= kResume) { sk = k - (kResume - kSplit); kok = (sk < kReal); }
        else                   { sk = 0; kok = false; }
        float v = (nok && kok) ? src[(size_t)srow * Ksrc + sk] : 0.0f;
        dst[idx] = (_Float16)v;
    }
}

__global__ void lrnn_pack_bias(const float* __restrict__ b_in, const float* __restrict__ b_hp,
                               const float* __restrict__ b_ih, const float* __restrict__ b_hh,
                               const float* __restrict__ b_o1, const float* __restrict__ b_o2,
                               float* __restrict__ dst) {
    int i = blockIdx.x * blockDim.x + threadIdx.x;
    if (i >= BIAS_TOTAL) return;
    float v = 0.0f;
    if (i < 192)       { if (i < 180) v = b_in[i]; }
    else if (i < 384)  { int j = i - 192;  if (j < 180) v = b_hp[j]; }
    else if (i < 960)  { int j = i - 384;  int g = j / 192, m = j % 192; if (m < 180) v = b_ih[g * 180 + m]; }
    else if (i < 1536) { int j = i - 960;  int g = j / 192, m = j % 192; if (m < 180) v = b_hh[g * 180 + m]; }
    else if (i < 1728) { int j = i - 1536; if (j < 180) v = b_o1[j]; }
    else               { int j = i - 1728; v = b_o2[j]; }
    dst[i] = v;
}

// ---------------- per-wave LDS state ----------------
struct __align__(16) WaveLds {
    _Float16 xin[16 * SR_XIN];    // [phys8 | cur64 | zero-pad] -> K=128
    _Float16 xact[16 * SR_ACT];   // x = gelu(...)   K=192
    _Float16 hbuf[16 * SR_ACT];   // hidden state    K=192 (persists across t)
    _Float16 o1act[16 * SR_ACT];  // gelu(o1)        K=192
    _Float16 curh[16 * SR_CURH];  // cur as f16      K=64  (persists across t)
    float    curf[16 * SR_CURF];  // cur as f32 (update precision, persists)
};

// ---------------- main recurrent kernel ----------------
// One wave = one 16-row batch tile for the whole T=512 sequence.
__global__ __launch_bounds__(64) void lrnn_main(const float* __restrict__ phys,
                                                const float* __restrict__ latents,
                                                const _Float16* __restrict__ wf,
                                                const float* __restrict__ bp,
                                                float* __restrict__ out) {
    __shared__ WaveLds lds[2];
    const int lane = threadIdx.x & 31;
    const int wave = threadIdx.x >> 5;
    WaveLds& L = lds[wave];
    const int Mbase = (blockIdx.x * 2 + wave) * 16;

    const int cl = lane & 15;   // C-fragment column within tile
    const int cg = lane >> 4;   // C-fragment row group (rows r + 8*cg)

    // ---- init: zero xin pad cols [72,128), stage latents -> xin[0..63]/curf/curh
    for (int e = lane; e < 16 * 56; e += 32) {
        int row = e / 56, c = e % 56;
        L.xin[row * SR_XIN + 72 + c] = (_Float16)0.0f;
    }
    for (int e = lane; e < 16 * 64; e += 32) {
        int row = e >> 6, c = e & 63;
        float v = latents[(size_t)(Mbase + row) * LL + c];
        L.xin[row * SR_XIN + c]   = (_Float16)v;
        L.curf[row * SR_CURF + c] = v;
        L.curh[row * SR_CURH + c] = (_Float16)v;
    }

    // ---- h0 = latents @ W_hp^T + b_hp   (K=64)
    {
        v16h a0 = load_a(L.xin, SR_XIN, 0, lane);
        v16h a1 = load_a(L.xin, SR_XIN, 32, lane);
        for (int nt = 0; nt < NT_H; ++nt) {
            v8f acc = bcast8(bp[BO_HP + nt * 16 + cl]);
            acc = wmma_f16(a0, load_b(wf + OFF_WHP + (nt * KT_HP + 0) * FR, lane), acc);
            acc = wmma_f16(a1, load_b(wf + OFF_WHP + (nt * KT_HP + 1) * FR, lane), acc);
#pragma unroll
            for (int r = 0; r < 8; ++r)
                L.hbuf[(r + 8 * cg) * SR_ACT + nt * 16 + cl] = (_Float16)acc[r];
        }
    }

#pragma clang loop unroll(disable)
    for (int t = 0; t < TT; ++t) {
        // ---- stage [phys_t | cur] into xin
        for (int e = lane; e < 16 * 8; e += 32) {
            int row = e >> 3, p = e & 7;
            L.xin[row * SR_XIN + p] =
                (_Float16)phys[((size_t)(Mbase + row) * TT + t) * PP + p];
        }
        for (int e = lane; e < 16 * 64; e += 32) {
            int row = e >> 6, c = e & 63;
            L.xin[row * SR_XIN + 8 + c] = L.curh[row * SR_CURH + c];
        }

        // ---- phase 1: x = gelu(xin @ W_in^T + b_in)   (K=128 -> 192 out)
        {
            v16h ax[KT_X];
#pragma unroll
            for (int kt = 0; kt < KT_X; ++kt) ax[kt] = load_a(L.xin, SR_XIN, kt * 32, lane);
            for (int nt = 0; nt < NT_H; ++nt) {
                v8f acc = bcast8(bp[BO_IN + nt * 16 + cl]);
#pragma unroll
                for (int kt = 0; kt < KT_X; ++kt)
                    acc = wmma_f16(ax[kt], load_b(wf + OFF_WIN + (nt * KT_X + kt) * FR, lane), acc);
#pragma unroll
                for (int r = 0; r < 8; ++r)
                    L.xact[(r + 8 * cg) * SR_ACT + nt * 16 + cl] = (_Float16)gelu_erf(acc[r]);
            }
        }

        // ---- phase 2: GRU cell (gi = x@W_ih^T, gh = h@W_hh^T, gate math, h update)
        {
            v16h axa[KT_H], aha[KT_H];
#pragma unroll
            for (int kt = 0; kt < KT_H; ++kt) {
                axa[kt] = load_a(L.xact, SR_ACT, kt * 32, lane);
                aha[kt] = load_a(L.hbuf, SR_ACT, kt * 32, lane);
            }
            for (int j = 0; j < NT_H; ++j) {
                v8f ir = bcast8(bp[BO_IH + 0 * 192 + j * 16 + cl]);
                v8f iz = bcast8(bp[BO_IH + 1 * 192 + j * 16 + cl]);
                v8f in_ = bcast8(bp[BO_IH + 2 * 192 + j * 16 + cl]);
                v8f hr = bcast8(bp[BO_HH + 0 * 192 + j * 16 + cl]);
                v8f hz = bcast8(bp[BO_HH + 1 * 192 + j * 16 + cl]);
                v8f hn = bcast8(bp[BO_HH + 2 * 192 + j * 16 + cl]);
#pragma unroll
                for (int kt = 0; kt < KT_H; ++kt) {
                    ir  = wmma_f16(axa[kt], load_b(wf + OFF_WIH + ((0 * NT_H + j) * KT_H + kt) * FR, lane), ir);
                    iz  = wmma_f16(axa[kt], load_b(wf + OFF_WIH + ((1 * NT_H + j) * KT_H + kt) * FR, lane), iz);
                    in_ = wmma_f16(axa[kt], load_b(wf + OFF_WIH + ((2 * NT_H + j) * KT_H + kt) * FR, lane), in_);
                    hr  = wmma_f16(aha[kt], load_b(wf + OFF_WHH + ((0 * NT_H + j) * KT_H + kt) * FR, lane), hr);
                    hz  = wmma_f16(aha[kt], load_b(wf + OFF_WHH + ((1 * NT_H + j) * KT_H + kt) * FR, lane), hz);
                    hn  = wmma_f16(aha[kt], load_b(wf + OFF_WHH + ((2 * NT_H + j) * KT_H + kt) * FR, lane), hn);
                }
#pragma unroll
                for (int r = 0; r < 8; ++r) {
                    int off = (r + 8 * cg) * SR_ACT + j * 16 + cl;
                    float hold = (float)L.hbuf[off];
                    float rr = sigm(ir[r] + hr[r]);
                    float zz = sigm(iz[r] + hz[r]);
                    float nn = tanhf(in_[r] + rr * hn[r]);
                    L.hbuf[off] = (_Float16)((1.0f - zz) * nn + zz * hold);
                }
            }
        }

        // ---- phase 3: o1 = gelu([h_new | cur] @ W_o1^T + b_o1)   (K=256)
        {
            v16h ahc[KT_O1];
#pragma unroll
            for (int kt = 0; kt < 6; ++kt) ahc[kt] = load_a(L.hbuf, SR_ACT, kt * 32, lane);
            ahc[6] = load_a(L.curh, SR_CURH, 0, lane);
            ahc[7] = load_a(L.curh, SR_CURH, 32, lane);
            for (int nt = 0; nt < NT_H; ++nt) {
                v8f acc = bcast8(bp[BO_O1 + nt * 16 + cl]);
#pragma unroll
                for (int kt = 0; kt < KT_O1; ++kt)
                    acc = wmma_f16(ahc[kt], load_b(wf + OFF_WO1 + (nt * KT_O1 + kt) * FR, lane), acc);
#pragma unroll
                for (int r = 0; r < 8; ++r)
                    L.o1act[(r + 8 * cg) * SR_ACT + nt * 16 + cl] = (_Float16)gelu_erf(acc[r]);
            }
        }

        // ---- phase 4: delta = o1act @ W_o2^T + b_o2 ; cur = clip(cur+delta,0,1); emit
        {
            v16h ao[KT_H];
#pragma unroll
            for (int kt = 0; kt < KT_H; ++kt) ao[kt] = load_a(L.o1act, SR_ACT, kt * 32, lane);
            for (int nt = 0; nt < NT_L; ++nt) {
                v8f acc = bcast8(bp[BO_O2 + nt * 16 + cl]);
#pragma unroll
                for (int kt = 0; kt < KT_H; ++kt)
                    acc = wmma_f16(ao[kt], load_b(wf + OFF_WO2 + (nt * KT_H + kt) * FR, lane), acc);
#pragma unroll
                for (int r = 0; r < 8; ++r) {
                    int row = r + 8 * cg, col = nt * 16 + cl;
                    float nv = L.curf[row * SR_CURF + col] + acc[r];
                    nv = fminf(fmaxf(nv, 0.0f), 1.0f);
                    L.curf[row * SR_CURF + col] = nv;
                    L.curh[row * SR_CURH + col] = (_Float16)nv;
                    out[((size_t)(Mbase + row) * TT + t) * LL + col] = nv;
                }
            }
        }
    }
}

// ---------------- host entry ----------------
extern "C" void kernel_launch(void* const* d_in, const int* in_sizes, int n_in,
                              void* d_out, int out_size, void* d_ws, size_t ws_size,
                              hipStream_t stream) {
    const float* phys    = (const float*)d_in[0];
    const float* latents = (const float*)d_in[1];
    const float* W_in = (const float*)d_in[2];
    const float* b_in = (const float*)d_in[3];
    const float* W_hp = (const float*)d_in[4];
    const float* b_hp = (const float*)d_in[5];
    const float* W_ih = (const float*)d_in[6];
    const float* b_ih = (const float*)d_in[7];
    const float* W_hh = (const float*)d_in[8];
    const float* b_hh = (const float*)d_in[9];
    const float* W_o1 = (const float*)d_in[10];
    const float* b_o1 = (const float*)d_in[11];
    const float* W_o2 = (const float*)d_in[12];
    const float* b_o2 = (const float*)d_in[13];

    _Float16* wfrag = (_Float16*)d_ws;
    float* biasp = (float*)((char*)d_ws + (size_t)FRAG_TOTAL * 2);

    const int BIG = 1 << 30;
    auto pk = [&](const float* src, int off, int NT, int KT, int Ksrc, int nGroup,
                  int nReal, int kSplit, int kResume, int kReal) {
        int n = NT * KT * FR;
        lrnn_pack_w<<<(n + 255) / 256, 256, 0, stream>>>(src, wfrag + off, NT, KT, Ksrc,
                                                         nGroup, nReal, kSplit, kResume, kReal);
    };
    pk(W_in, OFF_WIN, NT_H,     KT_X,  PP + LL, 0,   HH, PP + LL, BIG, PP + LL);
    pk(W_hp, OFF_WHP, NT_H,     KT_HP, LL,      0,   HH, LL,      BIG, LL);
    pk(W_ih, OFF_WIH, 3 * NT_H, KT_H,  HH,      192, HH, HH,      BIG, HH);
    pk(W_hh, OFF_WHH, 3 * NT_H, KT_H,  HH,      192, HH, HH,      BIG, HH);
    pk(W_o1, OFF_WO1, NT_H,     KT_O1, HH + LL, 0,   HH, HH,      192, HH + LL);
    pk(W_o2, OFF_WO2, NT_L,     KT_H,  HH,      0,   LL, HH,      BIG, HH);
    lrnn_pack_bias<<<(BIAS_TOTAL + 255) / 256, 256, 0, stream>>>(b_in, b_hp, b_ih, b_hh,
                                                                 b_o1, b_o2, biasp);

    // 128 batch tiles of 16 rows; 2 waves per block -> 64 blocks x 64 threads.
    lrnn_main<<<BB / 32, 64, 0, stream>>>(phys, latents, wfrag, biasp, (float*)d_out);
}
